// ScaledDotProductAttention_90640989815095
// MI455X (gfx1250) — compile-verified, matrix operand order checked by reading, not run on hardware
//
#include <hip/hip_runtime.h>

typedef __attribute__((ext_vector_type(16))) _Float16     v16h;
typedef __attribute__((ext_vector_type(8)))  float        v8f;
typedef __attribute__((ext_vector_type(4)))  float        v4f;
typedef __attribute__((ext_vector_type(4)))  unsigned int v4u;
typedef __attribute__((ext_vector_type(8)))  int          v8i;
typedef __attribute__((ext_vector_type(4)))  int          v4i;

#define Bb 2
#define Hh 16
#define Tt 2048
#define Dd 64
#define QT 32            // q rows per workgroup
#define VCH 64           // V keys staged per TDM chunk
#define VSTR 66          // padded LDS row stride (floats) for V tile
#define SCALE 0.125f     // 1/sqrt(64)
#define NEGINF (-1e9f)

#ifndef __has_builtin
#define __has_builtin(x) 0
#endif
#if __has_builtin(__builtin_amdgcn_tensor_load_to_lds) && __has_builtin(__builtin_amdgcn_s_wait_tensorcnt)
#define USE_TDM 1
#else
#define USE_TDM 0
#endif

#if USE_TDM
// Issue one TDM DMA: 64(keys) x 64(floats) f32 tile of V -> LDS, with LDS padding
// of 2 DWORDs every 64 DWORDs so the LDS row stride is VSTR=66 floats.
// D# encoding per CDNA5 ISA 8.3/8.4 (2D tensor; groups 2/3 and the extra group
// are zero/disabled). This toolchain exposes the 6-arg builtin form.
__device__ __forceinline__ void tdm_load_vtile(const float* gsrc, unsigned lds_byte_off) {
    unsigned long long ga = (unsigned long long)(size_t)gsrc;
    v4u g0 = { 1u,                                   // count=1, user descriptor
               lds_byte_off,                         // lds_addr [63:32]
               (unsigned)(ga & 0xffffffffu),         // global_addr[31:0]
               (unsigned)((ga >> 32) & 0x1ffffffu)   // global_addr[56:32]
                 | (2u << 30) };                     // type=2 ("image")
    v8i g1 = { (int)((2u << 16)        // data_size = 4B
                   | (1u << 20)        // pad_enable
                   | (5u << 22)        // pad_interval: code 5 = every 64 DWORDs
                   | (1u << 25)),      // pad_amount:   code 1 = 2 DWORDs
               (int)(64u  << 16),      // tensor_dim0 = 64  (bits 79:48, low16)
               (int)(2048u << 16),     // tensor_dim1 = 2048 (bits 111:80, low16)
               (int)(64u  << 16),      // tile_dim0 = 64    (bits 127:112)
               64,                     // tile_dim1 = 64 keys (bits 143:128)
               64,                     // tensor_dim0_stride = 64 (bits 207:160)
               0, 0 };
    v4i z4 = { 0, 0, 0, 0 };
    v8i z8 = { 0, 0, 0, 0, 0, 0, 0, 0 };
    __builtin_amdgcn_tensor_load_to_lds(g0, g1, z4, z4, z8, 0);
}
#endif

// One workgroup = 4 wave32 = 128 threads, one 32-row q tile of one (b,h).
__global__ __launch_bounds__(128, 1)
void attn_fused_kernel(const float* __restrict__ Q,
                       const float* __restrict__ K,
                       const float* __restrict__ V,
                       const int*   __restrict__ mask,
                       const float* __restrict__ bias,
                       float* __restrict__ out,
                       float* __restrict__ attn)
{
    __shared__ float    sc[QT][Tt + 4];       // 262,656 B score strip
    __shared__ _Float16 qs[QT][Dd + 4];       //   4,352 B f16 Q tile
    __shared__ float    vt[2][VCH][VSTR];     //  33,792 B double-buffered V tile

    const int qb   = blockIdx.x * QT;
    const int h    = blockIdx.y;
    const int b    = blockIdx.z;
    const int bh   = b * Hh + h;
    const int tid  = threadIdx.x;
    const int wave = tid >> 5;
    const int lane = tid & 31;
    const int lrm  = lane & 15;               // row/col owned by this lane
    const int koff = (lane >= 16) ? 8 : 0;    // hi-half K offset (16-bit A/B layout)

    const float* Qp = Q + (size_t)(bh * Tt + qb) * Dd;
    const float* Kp = K + (size_t)bh * Tt * Dd;
    const float* Vp = V + (size_t)bh * Tt * Dd;

    // ---- load Q tile into LDS as f16 ----
    for (int i = tid; i < QT * Dd; i += 128) {
        int r = i >> 6, d = i & 63;
        qs[r][d] = (_Float16)Qp[r * Dd + d];
    }
    __syncthreads();

    // ---- Phase 1: S = (Q K^T)*scale + bias, masked -> LDS ----
    for (int cb = wave; cb < Tt / 16; cb += 4) {
        const int key = cb * 16 + lrm;
        const float* kr = Kp + (size_t)key * Dd + koff;
        v16h b0, b1;
#pragma unroll
        for (int j = 0; j < 8; ++j) {
            b0[j]     = (_Float16)kr[j];
            b0[j + 8] = (_Float16)kr[j + 16];
            b1[j]     = (_Float16)kr[32 + j];
            b1[j + 8] = (_Float16)kr[32 + j + 16];
        }
        const int mv = mask[b * Tt + key];
#pragma unroll
        for (int rb = 0; rb < 2; ++rb) {
            const _Float16* qr = &qs[rb * 16 + lrm][koff];
            v16h a0, a1;
#pragma unroll
            for (int j = 0; j < 8; ++j) {
                a0[j]     = qr[j];
                a0[j + 8] = qr[j + 16];
                a1[j]     = qr[32 + j];
                a1[j + 8] = qr[32 + j + 16];
            }
            v8f c = {};
            c = __builtin_amdgcn_wmma_f32_16x16x32_f16(false, a0, false, b0, (short)0, c, false, false);
            c = __builtin_amdgcn_wmma_f32_16x16x32_f16(false, a1, false, b1, (short)0, c, false, false);
#pragma unroll
            for (int r = 0; r < 8; ++r) {
                int   qrl = rb * 16 + r + koff;
                int   qrg = qb + qrl;
                float s   = c[r] * SCALE + bias[((size_t)h * Tt + qrg) * Tt + key];
                sc[qrl][cb * 16 + lrm] = mv ? s : NEGINF;
            }
        }
    }
    __syncthreads();

    // ---- Phase 2: row softmax (vectorized), NT stream-out of attn ----
    for (int r = wave * (QT / 4); r < (wave + 1) * (QT / 4); ++r) {
        float m = -3.4e38f;
        for (int c4 = lane * 4; c4 < Tt; c4 += 128) {
            v4f x = *(const v4f*)&sc[r][c4];
            m = fmaxf(m, fmaxf(fmaxf(x[0], x[1]), fmaxf(x[2], x[3])));
        }
#pragma unroll
        for (int off = 16; off; off >>= 1) m = fmaxf(m, __shfl_xor(m, off, 32));
        float s = 0.f;
        for (int c4 = lane * 4; c4 < Tt; c4 += 128) {
            v4f x = *(const v4f*)&sc[r][c4];
            v4f e;
#pragma unroll
            for (int j = 0; j < 4; ++j) { e[j] = __expf(x[j] - m); s += e[j]; }
            *(v4f*)&sc[r][c4] = e;
        }
#pragma unroll
        for (int off = 16; off; off >>= 1) s += __shfl_xor(s, off, 32);
        const float inv = 1.0f / s;
        float* arow = attn + ((size_t)bh * Tt + qb + r) * Tt;
        for (int c4 = lane * 4; c4 < Tt; c4 += 128) {
            v4f p = *(const v4f*)&sc[r][c4];
            p = p * inv;
            *(v4f*)&sc[r][c4] = p;                               // keep for P.V
            __builtin_nontemporal_store(p, (v4f*)(arow + c4));   // NT stream-out
        }
    }
    __syncthreads();

    // ---- Phase 3: O = P V. Wave w owns D-columns [w*16, w*16+16) for both
    //      q row-blocks, so one B operand feeds two WMMAs. V is staged via TDM.
    const int dcol = wave * 16 + lrm;
    v8f acc0 = {}, acc1 = {};

#if USE_TDM
    if (wave == 0) tdm_load_vtile(Vp, (unsigned)(size_t)&vt[0][0][0]);
#endif
    for (int ch = 0; ch < Tt / VCH; ++ch) {
        const int buf = ch & 1;
#if USE_TDM
        if (wave == 0) {
            if (ch + 1 < Tt / VCH) {
                tdm_load_vtile(Vp + (size_t)(ch + 1) * VCH * Dd,
                               (unsigned)(size_t)&vt[buf ^ 1][0][0]);
                __builtin_amdgcn_s_wait_tensorcnt(1);  // prev chunk done, next in flight
            } else {
                __builtin_amdgcn_s_wait_tensorcnt(0);  // last chunk done
            }
        }
#else
        for (int i4 = tid; i4 < VCH * Dd / 4; i4 += 128) {
            int key = i4 >> 4, d = (i4 & 15) * 4;
            v4f w = *(const v4f*)(Vp + ((size_t)ch * VCH + key) * Dd + d);
            *(v4f*)&vt[buf][key][d] = w;
        }
#endif
        __syncthreads();   // V chunk ready (TDM complete / copies done)
#pragma unroll
        for (int ss = 0; ss < 2; ++ss) {   // two 32-key sub-steps per chunk
            const int kk = ch * VCH + ss * 32;
            const float* pr0 = &sc[0 * 16 + lrm][kk + koff];
            const float* pr1 = &sc[1 * 16 + lrm][kk + koff];
            v16h a0, a1, bv;
#pragma unroll
            for (int j = 0; j < 8; ++j) {
                const int kl = ss * 32 + koff + j;
                a0[j]     = (_Float16)pr0[j];
                a0[j + 8] = (_Float16)pr0[j + 16];
                a1[j]     = (_Float16)pr1[j];
                a1[j + 8] = (_Float16)pr1[j + 16];
                bv[j]     = (_Float16)vt[buf][kl][dcol];
                bv[j + 8] = (_Float16)vt[buf][kl + 16][dcol];
            }
            acc0 = __builtin_amdgcn_wmma_f32_16x16x32_f16(false, a0, false, bv, (short)0, acc0, false, false);
            acc1 = __builtin_amdgcn_wmma_f32_16x16x32_f16(false, a1, false, bv, (short)0, acc1, false, false);
        }
        __syncthreads();   // all waves done with vt[buf] before DMA overwrites it
    }
#pragma unroll
    for (int r = 0; r < 8; ++r) {
        int q0 = qb + r + koff;
        int q1 = qb + 16 + r + koff;
        out[((size_t)bh * Tt + q0) * Dd + dcol] = acc0[r];
        out[((size_t)bh * Tt + q1) * Dd + dcol] = acc1[r];
    }
}

extern "C" void kernel_launch(void* const* d_in, const int* in_sizes, int n_in,
                              void* d_out, int out_size, void* d_ws, size_t ws_size,
                              hipStream_t stream) {
    (void)in_sizes; (void)n_in; (void)out_size; (void)d_ws; (void)ws_size;
    const float* q    = (const float*)d_in[0];
    const float* k    = (const float*)d_in[1];
    const float* v    = (const float*)d_in[2];
    const int*   mask = (const int*)  d_in[3];
    const float* bias = (const float*)d_in[4];

    float* out  = (float*)d_out;                          // (B,H,T,D) first
    float* attn = out + (size_t)Bb * Hh * Tt * Dd;        // then (B,H,T,T)

    dim3 grid(Tt / QT, Hh, Bb);   // 64 x 16 x 2 = 2048 workgroups
    attn_fused_kernel<<<grid, 128, 0, stream>>>(q, k, v, mask, bias, out, attn);
}
